// RopeAttention_54365696033250
// MI455X (gfx1250) — matrix operand order repaired and checked by reading, not compiled, not run on hardware
//
#include <hip/hip_runtime.h>
#include <math.h>

// ---------------------------------------------------------------------------
// RoPE causal attention for MI455X (gfx1250, wave32, WMMA bf16 16x16x32).
// B=4, S=2048, STATE=1024, H=16, D=64, N_ROT=32.
// ---------------------------------------------------------------------------

typedef __attribute__((ext_vector_type(16))) __bf16 bf16x16;
typedef __attribute__((ext_vector_type(8)))  __bf16 bf16x8;
typedef __attribute__((ext_vector_type(8)))  float  floatx8;

union FragA { bf16x16 v; bf16x8 h[2]; };

constexpr int Bb   = 4;
constexpr int Ss   = 2048;
constexpr int Hh   = 16;
constexpr int Dd   = 64;
constexpr int Kdim = 1024;           // STATE == H*D
constexpr int Ndim = 1024;
constexpr int Mdim = Bb * Ss;        // 8192
constexpr int LDT  = 40;             // padded LDS row stride (bf16 elems), conflict-free

#define DEV __device__ __forceinline__

DEV unsigned short f2bf(float f) {
  unsigned u = __float_as_uint(f);
  unsigned r = u + 0x7FFFu + ((u >> 16) & 1u);   // round-to-nearest-even
  return (unsigned short)(r >> 16);
}
DEV float bf2f(unsigned short h) { return __uint_as_float(((unsigned)h) << 16); }
DEV unsigned pk2(float a, float b) { return (unsigned)f2bf(a) | ((unsigned)f2bf(b) << 16); }

DEV floatx8 fzero() { floatx8 z = {0.f,0.f,0.f,0.f,0.f,0.f,0.f,0.f}; return z; }

DEV floatx8 wmma_bf16(const FragA& a, const FragA& b, floatx8 c) {
  return __builtin_amdgcn_wmma_f32_16x16x32_bf16(false, a.v, false, b.v,
                                                 (short)0, c, false, false);
}

// 16-lane butterfly reductions in pure VALU via DPP16.
// masks {1,2,7,15} generate the full 16-lane xor group.
template <int CTRL>
DEV float dpp_shuf(float v) {
  int s = __float_as_int(v);
  return __int_as_float(__builtin_amdgcn_update_dpp(s, s, CTRL, 0xF, 0xF, true));
}
DEV float red_max16(float v) {
  v = fmaxf(v, dpp_shuf<0xB1>(v));   // quad_perm [1,0,3,2]  (xor 1)
  v = fmaxf(v, dpp_shuf<0x4E>(v));   // quad_perm [2,3,0,1]  (xor 2)
  v = fmaxf(v, dpp_shuf<0x141>(v));  // row_half_mirror      (xor 7)
  v = fmaxf(v, dpp_shuf<0x140>(v));  // row_mirror           (xor 15)
  return v;
}
DEV float red_sum16(float v) {
  v += dpp_shuf<0xB1>(v);
  v += dpp_shuf<0x4E>(v);
  v += dpp_shuf<0x141>(v);
  v += dpp_shuf<0x140>(v);
  return v;
}

// ---------------------------------------------------------------------------
// Weight fp32 -> bf16 (with optional folded scale). 4 elem/thread.
// ---------------------------------------------------------------------------
__global__ __launch_bounds__(256)
void convert_w(const float* __restrict__ src, unsigned short* __restrict__ dst,
               float scale) {
  int i = (blockIdx.x * 256 + threadIdx.x) * 4;
  float4 f = *(const float4*)(src + i);
  uint2 o;
  o.x = pk2(f.x * scale, f.y * scale);
  o.y = pk2(f.z * scale, f.w * scale);
  *(uint2*)(dst + i) = o;
}

// ---------------------------------------------------------------------------
// Tiled GEMM: Y[M,N] = A[M,K] * W[N,K]^T, 128x128 block, 8 waves of 32x64.
// Double-buffered LDS (one barrier per K-step; global loads overlap WMMA).
// AFMT: 0 = A fp32 global (convert to bf16 while staging), 1 = A bf16 global.
// EPI : 0 = store bf16 scattered to [B,H,S,D];  1 = store f32 row-major [M,N].
// ---------------------------------------------------------------------------
template <int AFMT, int EPI>
__global__ __launch_bounds__(256)
void gemm128(const void* __restrict__ Ag, const unsigned short* __restrict__ Wg,
             void* __restrict__ Og) {
  __shared__ __align__(16) unsigned short Al[2][128 * LDT];
  __shared__ __align__(16) unsigned short Bl[2][128 * LDT];

  const int tid  = threadIdx.x;
  const int wave = tid >> 5, lane = tid & 31;
  const int ln = lane & 15, lh = lane >> 4;
  const int nBase = blockIdx.x * 128, mBase = blockIdx.y * 128;
  const int wm = (wave >> 1) * 32;           // wave M offset (4 waves down)
  const int wn = (wave & 1) * 64;            // wave N offset (2 waves across)

  floatx8 acc[2][4];
  for (int mf = 0; mf < 2; ++mf)
    for (int nf = 0; nf < 4; ++nf) acc[mf][nf] = fzero();

  const int arow = tid >> 1, ahalf = (tid & 1) * 16;
  const size_t aOff = (size_t)(mBase + arow) * Kdim + ahalf;
  const size_t bOff = (size_t)(nBase + arow) * Kdim + ahalf;

  float4 fa[4];      // staged A (fp32 source)
  uint4  ua[2];      // staged A (bf16 source)
  uint4  ub[2];      // staged B (always bf16)

  auto gload = [&](int kt) {
    if (AFMT == 0) {
      const float* Ap = (const float*)Ag + aOff + kt;
      fa[0] = *(const float4*)(Ap + 0);
      fa[1] = *(const float4*)(Ap + 4);
      fa[2] = *(const float4*)(Ap + 8);
      fa[3] = *(const float4*)(Ap + 12);
    } else {
      const unsigned short* Ap = (const unsigned short*)Ag + aOff + kt;
      ua[0] = *(const uint4*)(Ap + 0);
      ua[1] = *(const uint4*)(Ap + 8);
    }
    const unsigned short* Bp = Wg + bOff + kt;
    ub[0] = *(const uint4*)(Bp + 0);
    ub[1] = *(const uint4*)(Bp + 8);
  };
  auto lstore = [&](int buf) {
    if (AFMT == 0) {
      uint4 s0 = make_uint4(pk2(fa[0].x, fa[0].y), pk2(fa[0].z, fa[0].w),
                            pk2(fa[1].x, fa[1].y), pk2(fa[1].z, fa[1].w));
      uint4 s1 = make_uint4(pk2(fa[2].x, fa[2].y), pk2(fa[2].z, fa[2].w),
                            pk2(fa[3].x, fa[3].y), pk2(fa[3].z, fa[3].w));
      *(uint4*)&Al[buf][arow * LDT + ahalf]     = s0;
      *(uint4*)&Al[buf][arow * LDT + ahalf + 8] = s1;
    } else {
      *(uint4*)&Al[buf][arow * LDT + ahalf]     = ua[0];
      *(uint4*)&Al[buf][arow * LDT + ahalf + 8] = ua[1];
    }
    *(uint4*)&Bl[buf][arow * LDT + ahalf]     = ub[0];
    *(uint4*)&Bl[buf][arow * LDT + ahalf + 8] = ub[1];
  };

  gload(0);
  lstore(0);
  __syncthreads();

  for (int kt = 0; kt < Kdim; kt += 32) {
    const int  buf  = (kt >> 5) & 1;
    const bool more = (kt + 32) < Kdim;
    if (more) {
      gload(kt + 32);                               // overlaps WMMAs below
      if (kt + 64 < Kdim) {                         // global_prefetch_b8
        if (AFMT == 0)
          __builtin_prefetch((const float*)Ag + aOff + kt + 64, 0, 1);
        else
          __builtin_prefetch((const unsigned short*)Ag + aOff + kt + 64, 0, 1);
        __builtin_prefetch(Wg + bOff + kt + 64, 0, 1);
      }
    }

    FragA af[2], bfr[4];
    for (int mf = 0; mf < 2; ++mf) {                // A layout: K at 8h, 16+8h
      int base = (wm + mf * 16 + ln) * LDT + 8 * lh;
      af[mf].h[0] = *(const bf16x8*)&Al[buf][base];
      af[mf].h[1] = *(const bf16x8*)&Al[buf][base + 16];
    }
    for (int nf = 0; nf < 4; ++nf) {                // B layout: K at 8h, 16+8h
      int base = (wn + nf * 16 + ln) * LDT + 8 * lh;
      bfr[nf].h[0] = *(const bf16x8*)&Bl[buf][base];
      bfr[nf].h[1] = *(const bf16x8*)&Bl[buf][base + 16];
    }
    for (int mf = 0; mf < 2; ++mf)
      for (int nf = 0; nf < 4; ++nf)
        acc[mf][nf] = wmma_bf16(af[mf], bfr[nf], acc[mf][nf]);

    if (more) {
      lstore(buf ^ 1);
      __syncthreads();
    }
  }

  // ---- epilogue ----
  for (int mf = 0; mf < 2; ++mf)
    for (int nf = 0; nf < 4; ++nf)
      for (int r = 0; r < 8; ++r) {
        int m = mBase + wm + mf * 16 + r + 8 * lh;
        int n = nBase + wn + nf * 16 + ln;
        float v = acc[mf][nf][r];
        if (EPI == 0) {
          int b = m >> 11, s = m & (Ss - 1);
          int h = n >> 6,  d = n & (Dd - 1);
          ((unsigned short*)Og)[(((size_t)(b * Hh + h)) * Ss + s) * Dd + d] = f2bf(v);
        } else {
          ((float*)Og)[(size_t)m * Ndim + n] = v;
        }
      }
}

// ---------------------------------------------------------------------------
// In-place RoPE on first 32 dims of Q and K (layout [B,H,S,D], bf16).
// ---------------------------------------------------------------------------
__global__ __launch_bounds__(256)
void rope_qk(unsigned short* __restrict__ Q, unsigned short* __restrict__ K) {
  const int tid = blockIdx.x * 256 + threadIdx.x;
  const int i  = tid & 15;
  const int s  = (tid >> 4) & (Ss - 1);
  const int bh = tid >> 15;
  const size_t base = ((size_t)bh * Ss + s) * Dd;
  const float invf = __expf(-(float)i * (9.210340371976184f / 16.0f)); // 10000^(-i/16)
  const float ang  = (float)s * invf;
  float sn, cs;
  __sincosf(ang, &sn, &cs);
  {
    float x1 = bf2f(Q[base + i]), x2 = bf2f(Q[base + i + 16]);
    Q[base + i]      = f2bf(x1 * cs - x2 * sn);
    Q[base + i + 16] = f2bf(x2 * cs + x1 * sn);
  }
  {
    float x1 = bf2f(K[base + i]), x2 = bf2f(K[base + i + 16]);
    K[base + i]      = f2bf(x1 * cs - x2 * sn);
    K[base + i + 16] = f2bf(x2 * cs + x1 * sn);
  }
}

// ---------------------------------------------------------------------------
// Flash attention, causal. Block = one (b,h) and 128 query rows; 8 waves x 16.
// Double-buffered K/V tiles; K tile staged with global_load_async_to_lds_b128
// (ASYNCcnt), V transposed through VGPRs during staging.
// Q,K,V: [B,H,S,D] bf16 (qk scale folded into weights).  Out: [B,S,H,D] bf16.
// ---------------------------------------------------------------------------
__global__ __launch_bounds__(256)
void attn_fa(const unsigned short* __restrict__ Qh,
             const unsigned short* __restrict__ Kh,
             const unsigned short* __restrict__ Vh,
             unsigned short* __restrict__ Og) {
  __shared__ __align__(16) unsigned short Kl[2][32 * 72];   // keys row-major, padded
  __shared__ __align__(16) unsigned short Vt[2][64 * 40];   // V transposed [dim][key]
  __shared__ __align__(16) unsigned short Pl[8 * 16 * 40];  // per-wave P scratch

  const int tid  = threadIdx.x;
  const int wave = tid >> 5, lane = tid & 31;
  const int ln = lane & 15, lh = lane >> 4;
  const int qTile = blockIdx.x & 15;       // S/128 = 16
  const int bh    = blockIdx.x >> 4;       // b*H + h
  const size_t hOff = (size_t)bh * Ss * Dd;
  const int qBase = qTile * 128;
  const int qw    = qBase + wave * 16;     // wave's first query row

  // Q fragments (A-matrix), resident for whole block
  FragA qf[2];
  {
    const unsigned short* qp = Qh + hOff + (size_t)(qw + ln) * Dd;
    for (int kc = 0; kc < 2; ++kc) {
      qf[kc].h[0] = *(const bf16x8*)(qp + kc * 32 + 8 * lh);
      qf[kc].h[1] = *(const bf16x8*)(qp + kc * 32 + 16 + 8 * lh);
    }
  }

  floatx8 o[4];
  for (int nf = 0; nf < 4; ++nf) o[nf] = fzero();
  float mrow[8], lrow[8];
  for (int r = 0; r < 8; ++r) { mrow[r] = -3.0e38f; lrow[r] = 0.f; }
  unsigned short* myP = Pl + wave * 16 * 40;

  const int srow = tid >> 3;               // staging: key row 0..31
  const int schk = (tid & 7) * 8;          // staging: 8-elem chunk
  const unsigned short* kRow = Kh + hOff + (size_t)srow * Dd + schk;
  const unsigned short* vRow = Vh + hOff + (size_t)srow * Dd + schk;
  const int nJ = qBase / 32 + 4;           // causal: keys up to qBase+127

  uint4 vreg;
  auto kAsync = [&](int j0, int buf) {     // 16B/lane global -> LDS, ASYNCcnt
    unsigned lds32 = (unsigned)(unsigned long long)(const void*)
                     &Kl[buf][srow * 72 + schk];
    const unsigned short* g = kRow + (size_t)j0 * Dd;
    asm volatile("global_load_async_to_lds_b128 %0, %1, off"
                 :: "v"(lds32), "v"(g) : "memory");
  };
  auto vLoad  = [&](int j0)  { vreg = *(const uint4*)(vRow + (size_t)j0 * Dd); };
  auto vStore = [&](int buf) {
    const unsigned short* ve = (const unsigned short*)&vreg;
#pragma unroll
    for (int e = 0; e < 8; ++e) Vt[buf][(schk + e) * 40 + srow] = ve[e];
  };

  kAsync(0, 0); vLoad(0); vStore(0);
  asm volatile("s_wait_asynccnt 0x0" ::: "memory");
  __syncthreads();

  for (int jt = 0; jt < nJ; ++jt) {
    const int  j0   = jt * 32;
    const int  buf  = jt & 1;
    const bool more = (jt + 1) < nJ;
    if (more) { kAsync(j0 + 32, buf ^ 1); vLoad(j0 + 32); }

    // ---- scores: two 16x16 tiles over 32 keys, K-dim = 64 (2 chained WMMAs) ----
    floatx8 sc0 = fzero(), sc1 = fzero();
    for (int kc = 0; kc < 2; ++kc) {
      FragA kf;
      int b0 = ln * 72 + kc * 32 + 8 * lh;             // B layout: K at 8h, 16+8h
      kf.h[0] = *(const bf16x8*)&Kl[buf][b0];
      kf.h[1] = *(const bf16x8*)&Kl[buf][b0 + 16];
      sc0 = wmma_bf16(qf[kc], kf, sc0);
      int b1 = (16 + ln) * 72 + kc * 32 + 8 * lh;
      kf.h[0] = *(const bf16x8*)&Kl[buf][b1];
      kf.h[1] = *(const bf16x8*)&Kl[buf][b1 + 16];
      sc1 = wmma_bf16(qf[kc], kf, sc1);
    }

    // ---- causal mask + online softmax (row = r + 8*lh, col = ln), DPP reductions ----
    float alpha[8];
#pragma unroll
    for (int r = 0; r < 8; ++r) {
      const int qrow = qw + r + 8 * lh;
      float s0 = sc0[r]; if (j0 + ln > qrow)      s0 = -3.0e38f;
      float s1 = sc1[r]; if (j0 + 16 + ln > qrow) s1 = -3.0e38f;
      const float mx   = red_max16(fmaxf(s0, s1));
      const float mnew = fmaxf(mrow[r], mx);
      alpha[r] = __expf(mrow[r] - mnew);
      const float p0 = __expf(s0 - mnew);
      const float p1 = __expf(s1 - mnew);
      const float rs = red_sum16(p0 + p1);
      lrow[r] = lrow[r] * alpha[r] + rs;
      mrow[r] = mnew;
      myP[(r + 8 * lh) * 40 + ln]      = f2bf(p0);
      myP[(r + 8 * lh) * 40 + 16 + ln] = f2bf(p1);
    }
#pragma unroll
    for (int nf = 0; nf < 4; ++nf)
#pragma unroll
      for (int r = 0; r < 8; ++r) o[nf][r] *= alpha[r];

    // P round-trip through per-wave LDS: wait for same-wave DS stores
    asm volatile("s_wait_dscnt 0" ::: "memory");

    // ---- P(16x32) x V(32x64): 4 accumulating WMMAs ----
    FragA pf;
    {
      int base = ln * 40 + 8 * lh;                     // A layout
      pf.h[0] = *(const bf16x8*)&myP[base];
      pf.h[1] = *(const bf16x8*)&myP[base + 16];
    }
    for (int nf = 0; nf < 4; ++nf) {
      FragA vf;
      int base = (nf * 16 + ln) * 40 + 8 * lh;         // B layout from Vt[dim][key]
      vf.h[0] = *(const bf16x8*)&Vt[buf][base];
      vf.h[1] = *(const bf16x8*)&Vt[buf][base + 16];
      o[nf] = wmma_bf16(pf, vf, o[nf]);
    }

    if (more) {
      vStore(buf ^ 1);
      asm volatile("s_wait_asynccnt 0x0" ::: "memory");
      __syncthreads();
    }
  }

  // ---- normalize and store to [B,S,H,D] bf16 ----
  const int b = bh >> 4, h = bh & 15;
#pragma unroll
  for (int r = 0; r < 8; ++r) {
    const float inv = 1.0f / lrow[r];
    const int s = qw + r + 8 * lh;
    for (int nf = 0; nf < 4; ++nf) {
      const int d = nf * 16 + ln;
      Og[(((size_t)b * Ss + s) * Hh + h) * Dd + d] = f2bf(o[nf][r] * inv);
    }
  }
}

// ---------------------------------------------------------------------------
// Host launcher
// ---------------------------------------------------------------------------
extern "C" void kernel_launch(void* const* d_in, const int* in_sizes, int n_in,
                              void* d_out, int out_size, void* d_ws, size_t ws_size,
                              hipStream_t stream) {
  (void)in_sizes; (void)n_in; (void)out_size; (void)ws_size;
  const float* q_src = (const float*)d_in[0];
  const float* k_src = (const float*)d_in[1];
  const float* v_src = (const float*)d_in[2];
  // d_in[3] = position_mask: causal, implemented analytically in-kernel.
  const float* Wq = (const float*)d_in[4];
  const float* Wk = (const float*)d_in[5];
  const float* Wv = (const float*)d_in[6];
  const float* Wo = (const float*)d_in[7];
  float* out = (float*)d_out;

  char* ws = (char*)d_ws;
  size_t off = 0;
  auto carve = [&](size_t bytes) -> void* {
    void* p = ws + off;
    off += (bytes + 255) & ~(size_t)255;
    return p;
  };
  const size_t wbytes = (size_t)Kdim * Ndim * sizeof(unsigned short);
  const size_t tbytes = (size_t)Mdim * Ndim * sizeof(unsigned short);
  unsigned short* Wq_b = (unsigned short*)carve(wbytes);
  unsigned short* Wk_b = (unsigned short*)carve(wbytes);
  unsigned short* Wv_b = (unsigned short*)carve(wbytes);
  unsigned short* Wo_b = (unsigned short*)carve(wbytes);
  unsigned short* Qb   = (unsigned short*)carve(tbytes);  // [B,H,S,D]
  unsigned short* Kb   = (unsigned short*)carve(tbytes);
  unsigned short* Vb   = (unsigned short*)carve(tbytes);
  unsigned short* Ao   = (unsigned short*)carve(tbytes);  // [B,S,H,D]

  const float qk_scale = 0.35355339059327373f;  // (1/sqrt(64))^0.5, folded into Wq,Wk

  const int cgrid = (Kdim * Ndim) / (256 * 4);
  convert_w<<<cgrid, 256, 0, stream>>>(Wq, Wq_b, qk_scale);
  convert_w<<<cgrid, 256, 0, stream>>>(Wk, Wk_b, qk_scale);
  convert_w<<<cgrid, 256, 0, stream>>>(Wv, Wv_b, 1.0f);
  convert_w<<<cgrid, 256, 0, stream>>>(Wo, Wo_b, 1.0f);

  dim3 gg(Ndim / 128, Mdim / 128);
  gemm128<0, 0><<<gg, 256, 0, stream>>>(q_src, Wq_b, Qb);
  gemm128<0, 0><<<gg, 256, 0, stream>>>(k_src, Wk_b, Kb);
  gemm128<0, 0><<<gg, 256, 0, stream>>>(v_src, Wv_b, Vb);

  rope_qk<<<(Bb * Hh * Ss * 16) / 256, 256, 0, stream>>>(Qb, Kb);

  attn_fa<<<Bb * Hh * (Ss / 128), 256, 0, stream>>>(Qb, Kb, Vb, Ao);

  gemm128<1, 1><<<gg, 256, 0, stream>>>(Ao, Wo_b, out);
}
